// OptimizedSSMBlock_84722524881477
// MI455X (gfx1250) — matrix operand order repaired and checked by reading, not compiled, not run on hardware
//
#include <hip/hip_runtime.h>
#include <hip/hip_bf16.h>

// ---------------------------------------------------------------------------
// Linear SSM  h_t = A h_{t-1} + b u_t,  y_t = c.h_t  (H=1024, T=2048, B=16)
// Rewritten as causal convolution y = w * u with w_k = c^T A^k b.
// Krylov doubling: G_{2m} = [G_m, P^m G_m], P^{2m} = (P^m)^2, P = A^T.
// 11 apply GEMMs + 10 squarings (~26 GFLOP, all L2-resident) on the
// v_wmma_f32_16x16x32_bf16 path using bf16x3 (hi/lo split) for ~f32 accuracy.
// Workspace requirement: ~28.01 MB.
// ---------------------------------------------------------------------------

typedef __attribute__((ext_vector_type(16))) __bf16 v16bf;
typedef __attribute__((ext_vector_type(8)))  float  v8f;

#define HH   1024
#define TT   2048
#define NB   16

// ---------------------------------------------------------------------------
// P = A^T   (h' = h @ A.T in the reference == column-vector h' = A h, and the
// Krylov vectors evolve with P = A^T acting on c)
// ---------------------------------------------------------------------------
__global__ void transposeP_kernel(const float* __restrict__ A, float* __restrict__ P) {
  int gid = blockIdx.x * blockDim.x + threadIdx.x;   // 1M threads
  int i = gid >> 10, j = gid & 1023;
  P[(long)i * HH + j] = A[(long)j * HH + i];
}

__global__ void initG_kernel(const float* __restrict__ C, float* __restrict__ G) {
  int h = blockIdx.x * blockDim.x + threadIdx.x;
  if (h < HH) G[(long)h * TT] = C[h];               // G[:,0] = c
}

// ---------------------------------------------------------------------------
// Split f32 matrix into bf16 hi/lo fragment buffers laid out in the exact
// per-lane WMMA operand order (ISA 7.12.2), so the GEMM kernel issues plain
// contiguous 32B loads per lane per fragment.
//
// A-operand (16x32, MxK) lane layout: lanes 0-15 row=lane, K = e(0..7), e+8(16..23)
//                                     lanes 16-31 row=lane-16, K = e+8(8..15), e+16(24..31)
// B-operand (32x16, KxN) lane layout: lanes 0-15 col=lane, K = e(0..7), e+8(16..23)
//                                     lanes 16-31 col=lane-16, K = e+8(8..15), e+16(24..31)
// Buffer index: (((kt*tiles)+tile)*32 + lane)*16 + e
// ---------------------------------------------------------------------------
__global__ void splitA_kernel(const float* __restrict__ src,
                              __bf16* __restrict__ hi, __bf16* __restrict__ lo,
                              int mtiles, int ld) {
  int gid  = blockIdx.x * blockDim.x + threadIdx.x;  // (kt*mtiles+it)*32+lane
  int lane = gid & 31;
  int tile = gid >> 5;
  int it = tile % mtiles;
  int kt = tile / mtiles;
  int row   = it * 16 + (lane & 15);
  int kbase = kt * 32 + ((lane >> 4) << 3);
  long base = (long)gid * 16;
#pragma unroll
  for (int e = 0; e < 16; ++e) {
    int k = kbase + (e & 7) + ((e >> 3) << 4);
    float x  = src[(long)row * ld + k];
    __bf16 h = (__bf16)x;
    float  r = x - (float)h;
    hi[base + e] = h;
    lo[base + e] = (__bf16)r;
  }
}

__global__ void splitB_kernel(const float* __restrict__ src,
                              __bf16* __restrict__ hi, __bf16* __restrict__ lo,
                              int ntiles, int ld, int nvalid) {
  int gid  = blockIdx.x * blockDim.x + threadIdx.x;  // (kt*ntiles+jt)*32+lane
  int lane = gid & 31;
  int tile = gid >> 5;
  int jt = tile % ntiles;
  int kt = tile / ntiles;
  int col   = jt * 16 + (lane & 15);
  int kbase = kt * 32 + ((lane >> 4) << 3);
  long base = (long)gid * 16;
  bool ok = (col < nvalid);
#pragma unroll
  for (int e = 0; e < 16; ++e) {
    int k = kbase + (e & 7) + ((e >> 3) << 4);
    float x  = ok ? src[(long)k * ld + col] : 0.0f;
    __bf16 h = (__bf16)x;
    float  r = x - (float)h;
    hi[base + e] = h;
    lo[base + e] = (__bf16)r;
  }
}

// ---------------------------------------------------------------------------
// bf16x3 WMMA GEMM: out(M=1024 x N) = Afrag(1024x1024) * Bfrag(1024xN)
// Each wave computes a 32x64 tile = 2x4 WMMA 16x16 tiles; 24 v_wmma per
// K=32 step (aL*bH + aH*bL + aH*bH accumulated in f32).
// Store is masked by colLimit and offset by colOff (writes G[:, m..2m-1]).
// ---------------------------------------------------------------------------
__global__ void __launch_bounds__(128)
wmma_gemm_bf16x3(const __bf16* __restrict__ aHi, const __bf16* __restrict__ aLo,
                 const __bf16* __restrict__ bHi, const __bf16* __restrict__ bLo,
                 float* __restrict__ out, int ldOut, int colOff, int colLimit,
                 int ntilesB) {
  const int mtilesA = HH / 16;       // 64
  const int KT      = HH / 32;       // 32 k-steps
  int wave = blockIdx.x * (blockDim.x >> 5) + (threadIdx.x >> 5);
  int lane = threadIdx.x & 31;
  int wm = wave & 31;                // 32 waves along M (32 rows each)
  int wn = wave >> 5;                // waves along N (64 cols each)

  v8f acc[2][4] = {};

  for (int kt = 0; kt < KT; ++kt) {
    v16bf aH[2], aL[2], bH[4], bL[4];
#pragma unroll
    for (int mi = 0; mi < 2; ++mi) {
      long idx = ((long)(kt * mtilesA + (wm * 2 + mi)) * 32 + lane) * 16;
      aH[mi] = *(const v16bf*)(aHi + idx);
      aL[mi] = *(const v16bf*)(aLo + idx);
    }
#pragma unroll
    for (int ni = 0; ni < 4; ++ni) {
      long idx = ((long)(kt * ntilesB + (wn * 4 + ni)) * 32 + lane) * 16;
      bH[ni] = *(const v16bf*)(bHi + idx);
      bL[ni] = *(const v16bf*)(bLo + idx);
    }
#pragma unroll
    for (int mi = 0; mi < 2; ++mi)
#pragma unroll
      for (int ni = 0; ni < 4; ++ni) {
        acc[mi][ni] = __builtin_amdgcn_wmma_f32_16x16x32_bf16(
            false, aL[mi], false, bH[ni], (short)0, acc[mi][ni], false, false);
        acc[mi][ni] = __builtin_amdgcn_wmma_f32_16x16x32_bf16(
            false, aH[mi], false, bL[ni], (short)0, acc[mi][ni], false, false);
        acc[mi][ni] = __builtin_amdgcn_wmma_f32_16x16x32_bf16(
            false, aH[mi], false, bH[ni], (short)0, acc[mi][ni], false, false);
      }
  }

  // C/D layout: VGPR v -> row v (+8 for lanes 16-31), col = lane & 15
  int nlo = lane & 15;
  int mhi = (lane >> 4) << 3;
#pragma unroll
  for (int mi = 0; mi < 2; ++mi)
#pragma unroll
    for (int ni = 0; ni < 4; ++ni) {
      int col = wn * 64 + ni * 16 + nlo;
      if (col < colLimit) {
#pragma unroll
        for (int v = 0; v < 8; ++v) {
          int row = wm * 32 + mi * 16 + mhi + v;
          out[(long)row * ldOut + colOff + col] = acc[mi][ni][v];
        }
      }
    }
}

// ---------------------------------------------------------------------------
// w[k] = sum_h G[h][k] * b[h]
// ---------------------------------------------------------------------------
__global__ void gemv_w_kernel(const float* __restrict__ G, const float* __restrict__ Bv,
                              float* __restrict__ w) {
  __shared__ float red[256];
  int k = blockIdx.x;
  float s = 0.0f;
  for (int h = threadIdx.x; h < HH; h += 256)
    s += G[(long)h * TT + k] * Bv[h];
  red[threadIdx.x] = s;
  __syncthreads();
  for (int st = 128; st > 0; st >>= 1) {
    if (threadIdx.x < st) red[threadIdx.x] += red[threadIdx.x + st];
    __syncthreads();
  }
  if (threadIdx.x == 0) w[k] = red[0];
}

// ---------------------------------------------------------------------------
// y[b][t] = D*u[b][t] + sum_{j<=t} w[t-j] * u[b][j]   (LDS-tiled, 34 M MACs)
// Block = (batch b, 256-wide t chunk); loop over j chunks with u-chunk and
// 511-wide w window staged in LDS.
// ---------------------------------------------------------------------------
__global__ void conv_causal_kernel(const float* __restrict__ u, const float* __restrict__ w,
                                   const float* __restrict__ Dv, float* __restrict__ y) {
  __shared__ float su[256];
  __shared__ float sw[512];
  int b   = blockIdx.x >> 3;        // TT/256 = 8 chunks per batch
  int tc  = blockIdx.x & 7;
  int tid = threadIdx.x;
  int t   = tc * 256 + tid;
  float acc = 0.0f;
  for (int jc = 0; jc <= tc; ++jc) {
    int base = (tc - jc) * 256;
    su[tid] = u[(long)b * TT + jc * 256 + tid];
    int i0 = base - 255;
    for (int idx = tid; idx < 512; idx += 256) {
      int wi = i0 + idx;
      sw[idx] = (wi >= 0 && wi < TT) ? w[wi] : 0.0f;
    }
    __syncthreads();
    if (jc < tc) {
      for (int jj = 0; jj < 256; ++jj)
        acc += sw[tid - jj + 255] * su[jj];
    } else {
      for (int jj = 0; jj <= tid; ++jj)
        acc += sw[tid - jj + 255] * su[jj];
    }
    __syncthreads();
  }
  y[(long)b * TT + t] = acc + Dv[0] * u[(long)b * TT + t];
}

// ---------------------------------------------------------------------------
extern "C" void kernel_launch(void* const* d_in, const int* in_sizes, int n_in,
                              void* d_out, int out_size, void* d_ws, size_t ws_size,
                              hipStream_t stream) {
  const float* u  = (const float*)d_in[0];   // (16,2048,1)
  const float* A  = (const float*)d_in[1];   // (1024,1024)
  const float* Bv = (const float*)d_in[2];   // (1024,1)
  const float* Cv = (const float*)d_in[3];   // (1,1024)
  const float* Dv = (const float*)d_in[4];   // (1,)
  float* y = (float*)d_out;                  // (16,2048,1)

  char* ws = (char*)d_ws;
  float*  P    = (float*)(ws);                       // 4 MB
  float*  Ptmp = (float*)(ws + (size_t)( 4 << 20));  // 4 MB
  float*  G    = (float*)(ws + (size_t)( 8 << 20));  // 8 MB (H x T)
  __bf16* PaHi = (__bf16*)(ws + (size_t)(16 << 20)); // 2 MB
  __bf16* PaLo = (__bf16*)(ws + (size_t)(18 << 20)); // 2 MB
  __bf16* PbHi = (__bf16*)(ws + (size_t)(20 << 20)); // 2 MB
  __bf16* PbLo = (__bf16*)(ws + (size_t)(22 << 20)); // 2 MB
  __bf16* GbHi = (__bf16*)(ws + (size_t)(24 << 20)); // 2 MB
  __bf16* GbLo = (__bf16*)(ws + (size_t)(26 << 20)); // 2 MB
  float*  wk   = (float*)(ws + (size_t)(28 << 20));  // 8 KB

  transposeP_kernel<<<(HH * HH) / 256, 256, 0, stream>>>(A, P);
  initG_kernel<<<HH / 256, 256, 0, stream>>>(Cv, G);

  float* Pcur = P;
  float* Pnext = Ptmp;
  for (int m = 1; m <= HH; m <<= 1) {
    int npad    = (m < 64) ? 64 : m;
    int ntilesB = npad / 16;
    int wavesN  = npad / 64;

    // fragment-order bf16 hi/lo splits (WMMA operand layouts)
    splitA_kernel<<<(32 * 64 * 32) / 256, 256, 0, stream>>>(Pcur, PaHi, PaLo, 64, HH);
    splitB_kernel<<<(32 * ntilesB * 32 + 255) / 256, 256, 0, stream>>>(G, GbHi, GbLo,
                                                                       ntilesB, TT, m);
    // G[:, m..2m-1] = P^m @ G[:, 0..m-1]
    wmma_gemm_bf16x3<<<8 * wavesN, 128, 0, stream>>>(PaHi, PaLo, GbHi, GbLo,
                                                     G, TT, m, m, ntilesB);
    if (m < HH) {
      // P^{2m} = (P^m)^2
      splitB_kernel<<<(32 * 64 * 32) / 256, 256, 0, stream>>>(Pcur, PbHi, PbLo, 64, HH, HH);
      wmma_gemm_bf16x3<<<8 * 16, 128, 0, stream>>>(PaHi, PaLo, PbHi, PbLo,
                                                   Pnext, HH, 0, HH, 64);
      float* tswap = Pcur; Pcur = Pnext; Pnext = tswap;
    }
  }

  gemv_w_kernel<<<TT, 256, 0, stream>>>(G, Bv, wk);
  conv_causal_kernel<<<NB * (TT / 256), 256, 0, stream>>>(u, wk, Dv, y);
}
